// SparseMoE_19061064859751
// MI455X (gfx1250) — matrix operand order repaired and checked by reading, not compiled, not run on hardware
//
#include <hip/hip_runtime.h>
#include <hip/hip_bf16.h>
#include <stdint.h>

// ---- problem constants (match reference setup_inputs) ----
#define T_TOK  4096   // B*S
#define DDIM   1024
#define EEXP   8
#define HDIM   4096
#define NPAIR  8192   // T_TOK * TOP_K (every token routes to exactly 2 experts)

typedef __attribute__((ext_vector_type(16))) __bf16 v16bf;
typedef __attribute__((ext_vector_type(8)))  float  v8f;

union FragU  { uint4 q[2]; v16bf v; };
union Pack32 { unsigned short h[32]; uint4 q[4]; };

__device__ __forceinline__ unsigned short f2bf(float f) {
  unsigned u = __float_as_uint(f);
  u += 0x7FFFu + ((u >> 16) & 1u);        // round-to-nearest-even
  return (unsigned short)(u >> 16);
}

// ---------------- init: zero counters ----------------
__global__ void moe_init_kernel(int* cnts) {
  if (threadIdx.x < 16) cnts[threadIdx.x] = 0;   // counts[8] + cursor[8]
}

// ---------------- router: logits, top-2, softmax ----------------
__global__ __launch_bounds__(256) void moe_router_kernel(
    const float* __restrict__ x, const float* __restrict__ Wr,
    const float* __restrict__ br,
    int* __restrict__ topidx, float* __restrict__ topgate,
    int* __restrict__ counts)
{
  const int t = blockIdx.x * 8 + (threadIdx.x >> 5);
  const int lane = threadIdx.x & 31;
  if (t >= T_TOK) return;
  float acc[EEXP];
  #pragma unroll
  for (int e = 0; e < EEXP; e++) acc[e] = 0.f;
  const float* xr = x + (size_t)t * DDIM;
  for (int d = lane; d < DDIM; d += 32) {
    const float xv = xr[d];
    const float* wr = Wr + (size_t)d * EEXP;
    #pragma unroll
    for (int e = 0; e < EEXP; e++) acc[e] = fmaf(xv, wr[e], acc[e]);
  }
  #pragma unroll
  for (int off = 16; off > 0; off >>= 1)
    #pragma unroll
    for (int e = 0; e < EEXP; e++) acc[e] += __shfl_xor(acc[e], off, 32);
  if (lane == 0) {
    #pragma unroll
    for (int e = 0; e < EEXP; e++) acc[e] += br[e];
    int e0 = 0;
    for (int e = 1; e < EEXP; e++) if (acc[e] > acc[e0]) e0 = e;
    int e1 = (e0 == 0) ? 1 : 0;
    for (int e = 0; e < EEXP; e++) if (e != e0 && acc[e] > acc[e1]) e1 = e;
    const float g0 = 1.f / (1.f + __expf(acc[e1] - acc[e0])); // 2-way softmax
    topidx[2*t]   = e0; topidx[2*t+1]  = e1;
    topgate[2*t]  = g0; topgate[2*t+1] = 1.f - g0;
    atomicAdd(&counts[e0], 1);
    atomicAdd(&counts[e1], 1);
  }
}

// ---------------- scan: exclusive prefix over 8 expert counts ----------------
__global__ void moe_scan_kernel(const int* __restrict__ counts,
                                int* __restrict__ offs, int* __restrict__ cursor) {
  if (threadIdx.x == 0) {
    int s = 0;
    for (int e = 0; e < EEXP; e++) { offs[e] = s; cursor[e] = s; s += counts[e]; }
  }
}

// ---------------- assign: compact pair rows grouped by expert ----------------
__global__ __launch_bounds__(256) void moe_assign_kernel(
    const int* __restrict__ topidx, int* __restrict__ cursor,
    int* __restrict__ rowTok, int* __restrict__ pairRow)
{
  const int t = blockIdx.x * 256 + threadIdx.x;
  if (t >= T_TOK) return;
  #pragma unroll
  for (int slot = 0; slot < 2; slot++) {
    const int e = topidx[2*t + slot];
    const int pos = atomicAdd(&cursor[e], 1);
    rowTok[pos] = t;
    pairRow[2*t + slot] = pos;
  }
}

// ---------------- weight transpose + f32->bf16: [E][K][N] -> [E][N][K] ----------------
__global__ __launch_bounds__(256) void moe_wtrans_kernel(
    const float* __restrict__ Wsrc, unsigned short* __restrict__ Wdst, int K, int N)
{
  __shared__ unsigned short tile[64][65];
  const int e  = blockIdx.z;
  const int k0 = blockIdx.y * 64;
  const int n0 = blockIdx.x * 64;
  const float* src = Wsrc + (size_t)e * K * N;
  unsigned short* dst = Wdst + (size_t)e * N * K;
  const int tx = threadIdx.x & 63;
  const int ty = threadIdx.x >> 6;          // 0..3
  #pragma unroll
  for (int i = 0; i < 16; i++) {
    const int k = ty + i * 4;
    tile[k][tx] = f2bf(src[(size_t)(k0 + k) * N + n0 + tx]);   // coalesced over n
  }
  __syncthreads();
  #pragma unroll
  for (int i = 0; i < 16; i++) {
    const int n = ty + i * 4;
    dst[(size_t)(n0 + n) * K + k0 + tx] = tile[tx][n];         // coalesced over k
  }
}

// ---------------- grouped GEMM, bf16 WMMA, f32 accumulate ----------------
// PASS 0: H = relu(x[rowTok] * W1[e] + b1[e]) -> bf16   (K=DDIM, N=HDIM)
// PASS 1: Y = H * W2[e] + b2[e]               -> f32    (K=HDIM, N=DDIM)
// BT 1: B pre-transposed bf16 [E][N][K]; BT 0: B raw f32 [E][K][N]
template<int PASS, int BT>
__device__ __forceinline__ void stage_fetch(
    Pack32& tw, int tid, int kk,
    const float* __restrict__ rowXf,          // PASS0 A: token row base (f32)
    const unsigned short* __restrict__ rowAb, // PASS1 A: pair row base (bf16)
    const float* __restrict__ colBf,          // BT0 B: column base (f32, stride N)
    const unsigned short* __restrict__ rowBt, // BT1 B: row base (bf16)
    int N)
{
  if (tid < 128) {
    if constexpr (PASS == 0) {
      const float4* s4 = (const float4*)(rowXf + kk);
      #pragma unroll
      for (int i = 0; i < 8; i++) {
        const float4 v = s4[i];
        tw.h[i*4+0] = f2bf(v.x); tw.h[i*4+1] = f2bf(v.y);
        tw.h[i*4+2] = f2bf(v.z); tw.h[i*4+3] = f2bf(v.w);
      }
    } else {
      const uint4* s4 = (const uint4*)(rowAb + kk);
      #pragma unroll
      for (int i = 0; i < 4; i++) tw.q[i] = s4[i];
    }
  } else {
    if constexpr (BT) {
      const uint4* s4 = (const uint4*)(rowBt + kk);
      #pragma unroll
      for (int i = 0; i < 4; i++) tw.q[i] = s4[i];
    } else {
      const float* src = colBf + (size_t)kk * N;
      #pragma unroll
      for (int k = 0; k < 32; k++) tw.h[k] = f2bf(src[(size_t)k * N]);
    }
  }
}

template<int PASS, int BT>
__global__ __launch_bounds__(256) void moe_gemm_kernel(
    const float* __restrict__ X,               // PASS0: tokens [T,K] f32
    const unsigned short* __restrict__ Abf,    // PASS1: [NPAIR,K] bf16
    const int* __restrict__ rowTok,            // pair row -> token (PASS0)
    const float* __restrict__ Wall,            // BT0: [E,K,N] f32
    const unsigned short* __restrict__ Wt,     // BT1: [E,N,K] bf16
    const float* __restrict__ ball,            // [E,N] f32
    const int* __restrict__ counts,
    const int* __restrict__ offs,
    unsigned short* __restrict__ outBF,        // PASS0 output
    float* __restrict__ outF,                  // PASS1 output
    int K, int N)
{
  const int e   = blockIdx.z;
  const int cnt = counts[e];
  const int mt  = blockIdx.y;
  if (mt * 128 >= cnt) return;                  // uniform per block: EXEC stays full
  const int goff   = offs[e];
  const int rem    = cnt - mt * 128;
  const int mvalid = rem < 128 ? rem : 128;
  const int nBase  = blockIdx.x * 128;

  __shared__ __align__(16) unsigned short As[2][128 * 40]; // [buf][row][k], pad 40
  __shared__ __align__(16) unsigned short Bs[2][128 * 40]; // [buf][n][k]

  const int tid  = threadIdx.x;
  const int lane = tid & 31;
  const int w    = tid >> 5;
  const int wm   = (w >> 1) * 32;   // wave tile rows: 0,32,64,96
  const int wn   = (w & 1) * 64;    // wave tile cols: 0,64

  // ---- loop-invariant staging pointers (hoists 64-bit address math) ----
  const float* rowXf = nullptr;
  const unsigned short* rowAb = nullptr;
  const float* colBf = nullptr;
  const unsigned short* rowBt = nullptr;
  if (tid < 128) {
    const int r  = tid;
    const int gr = (r < mvalid) ? r : (mvalid - 1);   // clamp; stores masked later
    if constexpr (PASS == 0)
      rowXf = X + (size_t)rowTok[goff + mt*128 + gr] * K;
    else
      rowAb = Abf + (size_t)(goff + mt*128 + gr) * K;
  } else {
    const int n = tid - 128;
    if constexpr (BT)
      rowBt = Wt + ((size_t)e * N + nBase + n) * K;
    else
      colBf = Wall + (size_t)e * K * N + nBase + n;
  }

  v8f acc[2][4];
  #pragma unroll
  for (int i = 0; i < 2; i++)
    #pragma unroll
    for (int j = 0; j < 4; j++)
      #pragma unroll
      for (int q = 0; q < 8; q++) acc[i][j][q] = 0.0f;

  Pack32 pre;
  stage_fetch<PASS, BT>(pre, tid, 0, rowXf, rowAb, colBf, rowBt, N);

  int buf = 0;
  for (int kk = 0; kk < K; kk += 32, buf ^= 1) {
    // ---- commit prefetched tile to LDS ----
    {
      uint4* dst = (uint4*)((tid < 128) ? (As[buf] + tid * 40)
                                        : (Bs[buf] + (tid - 128) * 40));
      #pragma unroll
      for (int i = 0; i < 4; i++) dst[i] = pre.q[i];
    }
    __syncthreads();

    // ---- issue next tile's global loads (overlap with WMMA below) ----
    const int kn = kk + 32;
    if (kn < K) {
      stage_fetch<PASS, BT>(pre, tid, kn, rowXf, rowAb, colBf, rowBt, N);
      const int kp = kn + 32;                 // depth-2 hint -> global_prefetch_b8
      if (kp < K) {
        if (tid < 128) {
          if constexpr (PASS == 0) __builtin_prefetch(rowXf + kp, 0, 1);
          else                     __builtin_prefetch(rowAb + kp, 0, 1);
        } else {
          if constexpr (BT)        __builtin_prefetch(rowBt + kp, 0, 1);
          else                     __builtin_prefetch(colBf + (size_t)kp * N, 0, 1);
        }
      }
    }

    // ---- build WMMA fragments (16-bit A 16x32 / B 32x16 lane layouts) ----
    const int rsel  = lane & 15;
    const int khalf = lane >> 4;
    FragU af[2], bfr[4];
    #pragma unroll
    for (int mf = 0; mf < 2; mf++) {
      const unsigned short* base = As[buf] + (wm + mf*16 + rsel) * 40;
      af[mf].q[0] = *(const uint4*)(base + khalf * 8);        // K 0..7 / 8..15
      af[mf].q[1] = *(const uint4*)(base + 16 + khalf * 8);   // K 16..23 / 24..31
    }
    #pragma unroll
    for (int nf = 0; nf < 4; nf++) {
      const unsigned short* base = Bs[buf] + (wn + nf*16 + rsel) * 40;
      bfr[nf].q[0] = *(const uint4*)(base + khalf * 8);       // K 0..7 / 8..15
      bfr[nf].q[1] = *(const uint4*)(base + 16 + khalf * 8);  // K 16..23 / 24..31
    }
    #pragma unroll
    for (int mf = 0; mf < 2; mf++)
      #pragma unroll
      for (int nf = 0; nf < 4; nf++)
        acc[mf][nf] = __builtin_amdgcn_wmma_f32_16x16x32_bf16(
            false, af[mf].v, false, bfr[nf].v,
            (short)0, acc[mf][nf], false, false);
    // no barrier here: next iteration writes the other LDS buffer; the
    // barrier at the top of the next iteration orders buffer reuse.
  }

  // ---- epilogue: bias (+relu/bf16 for PASS0) ----
  const float* bias = ball + (size_t)e * N;
  #pragma unroll
  for (int nf = 0; nf < 4; nf++) {
    const int col = nBase + wn + nf*16 + (lane & 15);
    const float bv = bias[col];
    #pragma unroll
    for (int mf = 0; mf < 2; mf++) {
      const int r0 = wm + mf*16 + (lane >> 4) * 8;
      #pragma unroll
      for (int j = 0; j < 8; j++) {
        const int rl = r0 + j;
        if (rl < mvalid) {
          float v = acc[mf][nf][j] + bv;
          const size_t row = (size_t)(goff + mt*128 + rl);
          if constexpr (PASS == 0) {
            v = v > 0.f ? v : 0.f;
            outBF[row * N + col] = f2bf(v);
          } else {
            outF[row * N + col] = v;
          }
        }
      }
    }
  }
}

// ---------------- combine: out[t] = g0*Y[p0] + g1*Y[p1] ----------------
__global__ __launch_bounds__(256) void moe_combine_kernel(
    const float* __restrict__ Y, const int* __restrict__ pairRow,
    const float* __restrict__ topgate, float* __restrict__ out)
{
  const int t = blockIdx.x;
  const int c = threadIdx.x;                  // 256 threads * float4 = 1024 = DDIM
  const int p0 = pairRow[2*t], p1 = pairRow[2*t+1];
  const float g0 = topgate[2*t], g1 = topgate[2*t+1];
  const float4 a = ((const float4*)(Y + (size_t)p0 * DDIM))[c];
  const float4 b = ((const float4*)(Y + (size_t)p1 * DDIM))[c];
  float4 o;
  o.x = g0*a.x + g1*b.x; o.y = g0*a.y + g1*b.y;
  o.z = g0*a.z + g1*b.z; o.w = g0*a.w + g1*b.w;
  ((float4*)(out + (size_t)t * DDIM))[c] = o;
}

extern "C" void kernel_launch(void* const* d_in, const int* in_sizes, int n_in,
                              void* d_out, int out_size, void* d_ws, size_t ws_size,
                              hipStream_t stream) {
  const float* x  = (const float*)d_in[0];  // [T, D]
  const float* Wr = (const float*)d_in[1];  // [D, E]
  const float* br = (const float*)d_in[2];  // [E]
  const float* W1 = (const float*)d_in[3];  // [E, D, H]
  const float* b1 = (const float*)d_in[4];  // [E, H]
  const float* W2 = (const float*)d_in[5];  // [E, H, D]
  const float* b2 = (const float*)d_in[6];  // [E, D]
  float* out = (float*)d_out;

  // ---- workspace layout ----
  char* w = (char*)d_ws;
  int*   counts  = (int*)w;               // [8]
  int*   cursor  = counts + 8;            // [8]
  int*   offs    = counts + 16;           // [8]
  int*   topidx  = (int*)(w + 256);                    // [2T]
  float* topgate = (float*)(w + 256 + 32768);          // [2T]
  int*   pairRow = (int*)(w + 256 + 65536);            // [2T]
  int*   rowTok  = (int*)(w + 256 + 98304);            // [NPAIR]
  const size_t big    = 1u << 18;                            // 256 KB
  const size_t HBYTES = (size_t)NPAIR * HDIM * 2;            // 64 MB
  const size_t YBYTES = (size_t)NPAIR * DDIM * 4;            // 32 MB
  const size_t WTB    = (size_t)EEXP * DDIM * HDIM * 2;      // 64 MB each
  unsigned short* Hbuf = (unsigned short*)(w + big);
  float*          Ybuf = (float*)(w + big + HBYTES);
  unsigned short* W1t  = (unsigned short*)(w + big + HBYTES + YBYTES);
  unsigned short* W2t  = (unsigned short*)(w + big + HBYTES + YBYTES + WTB);
  const bool useT = ws_size >= big + HBYTES + YBYTES + 2 * WTB;

  if (useT) {   // one-time f32->bf16 transpose of both weight tensors
    moe_wtrans_kernel<<<dim3(HDIM/64, DDIM/64, EEXP), 256, 0, stream>>>(W1, W1t, DDIM, HDIM);
    moe_wtrans_kernel<<<dim3(DDIM/64, HDIM/64, EEXP), 256, 0, stream>>>(W2, W2t, HDIM, DDIM);
  }

  moe_init_kernel<<<1, 32, 0, stream>>>(counts);
  moe_router_kernel<<<T_TOK / 8, 256, 0, stream>>>(x, Wr, br, topidx, topgate, counts);
  moe_scan_kernel<<<1, 32, 0, stream>>>(counts, offs, cursor);
  moe_assign_kernel<<<T_TOK / 256, 256, 0, stream>>>(topidx, cursor, rowTok, pairRow);

  const dim3 g1grid(HDIM / 128, T_TOK / 128, EEXP);
  const dim3 g2grid(DDIM / 128, T_TOK / 128, EEXP);
  if (useT) {
    moe_gemm_kernel<0, 1><<<g1grid, 256, 0, stream>>>(
        x, (const unsigned short*)nullptr, rowTok, (const float*)nullptr, W1t,
        b1, counts, offs, Hbuf, (float*)nullptr, DDIM, HDIM);
    moe_gemm_kernel<1, 1><<<g2grid, 256, 0, stream>>>(
        (const float*)nullptr, Hbuf, (const int*)nullptr, (const float*)nullptr, W2t,
        b2, counts, offs, (unsigned short*)nullptr, Ybuf, HDIM, DDIM);
  } else {      // fallback: stage B straight from f32 weights
    moe_gemm_kernel<0, 0><<<g1grid, 256, 0, stream>>>(
        x, (const unsigned short*)nullptr, rowTok, W1, (const unsigned short*)nullptr,
        b1, counts, offs, Hbuf, (float*)nullptr, DDIM, HDIM);
    moe_gemm_kernel<1, 0><<<g2grid, 256, 0, stream>>>(
        (const float*)nullptr, Hbuf, (const int*)nullptr, W2, (const unsigned short*)nullptr,
        b2, counts, offs, (unsigned short*)nullptr, Ybuf, HDIM, DDIM);
  }

  moe_combine_kernel<<<T_TOK, 256, 0, stream>>>(Ybuf, pairRow, topgate, out);

  (void)in_sizes; (void)n_in; (void)out_size; (void)ws_size;
}
